// TemporalModel_72155450573542
// MI455X (gfx1250) — matrix operand order repaired and checked by reading, not compiled, not run on hardware
//
#include <hip/hip_runtime.h>
#include <hip/hip_bf16.h>

#define HID   200
#define TLEN  4096
#define BSZ   2
#define KDIM  6400
#define NDIM  800   // 4*HID

// ---------- bf16 helpers (bit-level, RNE) ----------
static __device__ __forceinline__ unsigned short f2bf(float f) {
    unsigned int u = __float_as_uint(f);
    u += 0x7fffu + ((u >> 16) & 1u);       // round-to-nearest-even
    return (unsigned short)(u >> 16);
}
static __device__ __forceinline__ float bf_lo(unsigned int packed) {
    return __uint_as_float(packed << 16);
}
static __device__ __forceinline__ float bf_hi(unsigned int packed) {
    return __uint_as_float(packed & 0xffff0000u);
}
static __device__ __forceinline__ float sigf(float x) {
    return 1.f / (1.f + __expf(-x));
}

// ---------- WMMA types ----------
typedef __attribute__((ext_vector_type(16))) __bf16 bf16x16;
typedef __attribute__((ext_vector_type(8)))  float  f32x8;
union FragAB { bf16x16 v; uint4 q[2]; };
union Q4 { uint4 v; unsigned int d[4]; };

// =====================================================================
// Kernel 1: Conv2d(3->16, 3x3, pad=1) + ReLU, output bf16 feat[t*B+b][6400]
// =====================================================================
__global__ __launch_bounds__(256) void conv_relu_bf16_kernel(
        const float* __restrict__ vin, const float* __restrict__ cw,
        unsigned short* __restrict__ featb) {
    __shared__ float sin_[3 * 20 * 20];
    __shared__ float sw_[16 * 3 * 3 * 3];
    const int f = blockIdx.x;                     // frame index = b*T + t
    const float* src = vin + (size_t)f * 1200;
    for (int i = threadIdx.x; i < 1200; i += 256) sin_[i] = src[i];
    for (int i = threadIdx.x; i < 432;  i += 256) sw_[i]  = cw[i];
    __syncthreads();
    const int t = f & (TLEN - 1);
    const int b = f >> 12;
    const size_t orow = (size_t)(t * BSZ + b) * KDIM;  // [T,B,6400] row order
    for (int i = threadIdx.x; i < KDIM; i += 256) {
        const int oc = i / 400, pix = i % 400, y = pix / 20, x = pix % 20;
        float acc = 0.f;
        #pragma unroll
        for (int ic = 0; ic < 3; ic++) {
            #pragma unroll
            for (int ky = 0; ky < 3; ky++) {
                const int yy = y + ky - 1;
                if (yy < 0 || yy >= 20) continue;
                #pragma unroll
                for (int kx = 0; kx < 3; kx++) {
                    const int xx = x + kx - 1;
                    if (xx < 0 || xx >= 20) continue;
                    acc += sin_[ic * 400 + yy * 20 + xx] *
                           sw_[((oc * 3 + ic) * 3 + ky) * 3 + kx];
                }
            }
        }
        featb[orow + i] = f2bf(fmaxf(acc, 0.f));
    }
}

// =====================================================================
// Kernel 2: W_ih fp32 -> bf16, with gate-interleaved row permutation:
// original row r = g*200 + j  ->  permuted row p = 4*j + g
// =====================================================================
__global__ void cvt_permute_bf16_kernel(const float* __restrict__ w,
                                        unsigned short* __restrict__ wb) {
    const int i = blockIdx.x * blockDim.x + threadIdx.x;
    if (i < NDIM * KDIM) {
        const int r = i / KDIM, k = i - r * KDIM;
        const int p = (r % HID) * 4 + (r / HID);
        wb[(size_t)p * KDIM + k] = f2bf(w[i]);
    }
}

// =====================================================================
// Kernel 3: WMMA bf16 GEMM: xg[8192,800] = feat @ W_ih^T + bias
// (columns are gate-interleaved; bias looked up via inverse permutation)
// =====================================================================
__global__ __launch_bounds__(256, 2) void gemm_wmma_kernel(
        const unsigned short* __restrict__ A,   // feat bf16 [8192][6400]
        const unsigned short* __restrict__ Bw,  // W_ih bf16 [800][6400] permuted rows
        const float* __restrict__ bih, const float* __restrict__ bhh,
        float* __restrict__ xg) {
    const int lane = threadIdx.x & 31;
    const int wave = threadIdx.x >> 5;
    const int half = lane >> 4;
    const int l16  = lane & 15;
    const int rowbase = blockIdx.x * 256 + wave * 32;
    const int colb    = blockIdx.y * 64;

    f32x8 acc[2][4];
    #pragma unroll
    for (int i = 0; i < 2; i++)
        #pragma unroll
        for (int j = 0; j < 4; j++)
            acc[i][j] = (f32x8){0.f,0.f,0.f,0.f,0.f,0.f,0.f,0.f};

    for (int k0 = 0; k0 < KDIM; k0 += 32) {
        FragAB a[2];
        #pragma unroll
        for (int it = 0; it < 2; it++) {
            const unsigned short* ap = A + (size_t)(rowbase + it * 16 + l16) * KDIM;
            a[it].q[0] = *(const uint4*)(ap + k0 + half * 16);
            a[it].q[1] = *(const uint4*)(ap + k0 + half * 16 + 8);
        }
        FragAB bf[4];
        #pragma unroll
        for (int jt = 0; jt < 4; jt++) {
            int col = colb + jt * 16 + l16;
            col = col < NDIM ? col : NDIM - 1;    // clamp (stores are guarded)
            const unsigned short* bp = Bw + (size_t)col * KDIM + k0 + half * 16;
            bf[jt].q[0] = *(const uint4*)(bp);
            bf[jt].q[1] = *(const uint4*)(bp + 8);
        }
        #pragma unroll
        for (int it = 0; it < 2; it++)
            #pragma unroll
            for (int jt = 0; jt < 4; jt++)
                acc[it][jt] = __builtin_amdgcn_wmma_f32_16x16x32_bf16(
                    false, a[it].v, false, bf[jt].v,
                    (short)0, acc[it][jt], false, false);
    }

    #pragma unroll
    for (int it = 0; it < 2; it++) {
        #pragma unroll
        for (int jt = 0; jt < 4; jt++) {
            const int col = colb + jt * 16 + l16;      // permuted column
            if (col < NDIM) {
                const int orig = (col & 3) * HID + (col >> 2);
                const float bias = bih[orig] + bhh[orig];
                #pragma unroll
                for (int r = 0; r < 8; r++) {
                    const int row = rowbase + it * 16 + r + half * 8;
                    xg[(size_t)row * NDIM + col] = acc[it][jt][r] + bias;
                }
            }
        }
    }
}

// =====================================================================
// Kernel 4: persistent WMMA LSTM scan. 1 block x 800 threads (25 wave32).
// W_hh bf16 in LDS [800 permuted rows][200]; per step: gates = H @ Whh^T via
// 6x K32 WMMA chunks + 8-wide VALU tail; gate regroup via __shfl (cols are
// 4*j+gate interleaved); c in registers; 2 barriers/step.
// =====================================================================
__global__ __launch_bounds__(800, 1) void lstm_wmma_kernel(
        const float* __restrict__ xg,    // [T*B, 800] permuted cols, row = t*B+b
        const float* __restrict__ Whh,   // [800][200] fp32 (original order)
        float* __restrict__ out) {       // [T, B, 200]
    __shared__ unsigned short w6[NDIM * HID];  // 320,000 B  bf16, permuted rows
    __shared__ unsigned short hA[2 * 192];     //     768 B  h (k=0..191) bf16
    __shared__ float          ht[2 * 8];       //      64 B  h (k=192..199) f32

    const int tid  = threadIdx.x;
    const int lane = tid & 31, wave = tid >> 5;
    const int half = lane >> 4, l16 = lane & 15;
    const int n0 = wave * 2, n1 = wave * 2 + 1;          // N-tiles 0..49

    // ---- one-time init: permuted W_hh -> LDS bf16 ----
    {
        const int r = tid;                               // original row g*200+j
        const int p = (r % HID) * 4 + (r / HID);
        const float* src = Whh + (size_t)r * HID;
        unsigned short* dst = &w6[p * HID];
        for (int k = 0; k < HID; k++) dst[k] = f2bf(src[k]);
    }
    if (tid < 2 * 192) hA[tid] = 0;
    if (tid < 16) ht[tid] = 0.f;
    float c00 = 0.f, c01 = 0.f, c10 = 0.f, c11 = 0.f;    // c[tile][batch]
    __syncthreads();

    const int col0 = n0 * 16 + l16, col1 = n1 * 16 + l16;  // permuted cols
    const unsigned short* w6r0 = &w6[col0 * HID];
    const unsigned short* w6r1 = &w6[col1 * HID];
    const int j0 = n0 * 4 + (l16 >> 2);                  // hidden index, tile0
    const int j1 = n1 * 4 + (l16 >> 2);
    const bool writer = ((l16 & 3) == 0) && (lane < 16);
    const int base = lane & ~3;

    #pragma unroll 1
    for (int t = 0; t < TLEN; t++) {
        const size_t xrow = (size_t)t * 1600;
        const float xg00 = xg[xrow + col0];
        const float xg01 = xg[xrow + 800 + col0];
        const float xg10 = xg[xrow + col1];
        const float xg11 = xg[xrow + 800 + col1];

        // ---- A fragments (h): rows M=0,1 from LDS, rows 2..15 zero ----
        FragAB a[6];
        #pragma unroll
        for (int kc = 0; kc < 6; kc++) {
            const int k0 = kc * 32;
            if (l16 < 2) {
                a[kc].q[0] = *(const uint4*)&hA[l16 * 192 + k0 + half * 16];
                a[kc].q[1] = *(const uint4*)&hA[l16 * 192 + k0 + half * 16 + 8];
            } else {
                a[kc].q[0] = make_uint4(0u, 0u, 0u, 0u);
                a[kc].q[1] = make_uint4(0u, 0u, 0u, 0u);
            }
        }
        // ---- K tail (k=192..199), VALU; reads ht ----
        float t00 = 0.f, t01 = 0.f, t10 = 0.f, t11 = 0.f;
        {
            Q4 q0, q1;
            q0.v = *(const uint4*)(w6r0 + 192);
            q1.v = *(const uint4*)(w6r1 + 192);
            #pragma unroll
            for (int m = 0; m < 4; m++) {
                const float h0a = ht[2 * m], h0b = ht[2 * m + 1];
                const float h1a = ht[8 + 2 * m], h1b = ht[8 + 2 * m + 1];
                const float lo0 = bf_lo(q0.d[m]), hi0 = bf_hi(q0.d[m]);
                const float lo1 = bf_lo(q1.d[m]), hi1 = bf_hi(q1.d[m]);
                t00 += lo0 * h0a + hi0 * h0b;  t01 += lo0 * h1a + hi0 * h1b;
                t10 += lo1 * h0a + hi1 * h0b;  t11 += lo1 * h1a + hi1 * h1b;
            }
        }
        __syncthreads();          // all reads of hA/ht done before rewrites

        // ---- WMMA: 6 K-chunks x 2 N-tiles ----
        f32x8 acc0 = (f32x8){0.f,0.f,0.f,0.f,0.f,0.f,0.f,0.f};
        f32x8 acc1 = (f32x8){0.f,0.f,0.f,0.f,0.f,0.f,0.f,0.f};
        #pragma unroll
        for (int kc = 0; kc < 6; kc++) {
            const int k0 = kc * 32 + half * 16;
            FragAB b0f, b1f;
            b0f.q[0] = *(const uint4*)(w6r0 + k0);
            b0f.q[1] = *(const uint4*)(w6r0 + k0 + 8);
            b1f.q[0] = *(const uint4*)(w6r1 + k0);
            b1f.q[1] = *(const uint4*)(w6r1 + k0 + 8);
            acc0 = __builtin_amdgcn_wmma_f32_16x16x32_bf16(
                false, a[kc].v, false, b0f.v, (short)0, acc0, false, false);
            acc1 = __builtin_amdgcn_wmma_f32_16x16x32_bf16(
                false, a[kc].v, false, b1f.v, (short)0, acc1, false, false);
        }
        // gate values valid on lanes 0..15 (acc[0] = batch0 (M=0), acc[1] = batch1)
        const float gv00 = acc0[0] + t00 + xg00;
        const float gv01 = acc0[1] + t01 + xg01;
        const float gv10 = acc1[0] + t10 + xg10;
        const float gv11 = acc1[1] + t11 + xg11;

        // gather i,f,g,o within each group of 4 lanes (cols are 4*j+gate)
        const float i00 = __shfl(gv00, base + 0), f00 = __shfl(gv00, base + 1),
                    g00 = __shfl(gv00, base + 2), o00 = __shfl(gv00, base + 3);
        const float i01 = __shfl(gv01, base + 0), f01 = __shfl(gv01, base + 1),
                    g01 = __shfl(gv01, base + 2), o01 = __shfl(gv01, base + 3);
        const float i10 = __shfl(gv10, base + 0), f10 = __shfl(gv10, base + 1),
                    g10 = __shfl(gv10, base + 2), o10 = __shfl(gv10, base + 3);
        const float i11 = __shfl(gv11, base + 0), f11 = __shfl(gv11, base + 1),
                    g11 = __shfl(gv11, base + 2), o11 = __shfl(gv11, base + 3);

        if (writer) {
            c00 = sigf(f00) * c00 + sigf(i00) * tanhf(g00);
            c01 = sigf(f01) * c01 + sigf(i01) * tanhf(g01);
            c10 = sigf(f10) * c10 + sigf(i10) * tanhf(g10);
            c11 = sigf(f11) * c11 + sigf(i11) * tanhf(g11);
            const float h00 = sigf(o00) * tanhf(c00);
            const float h01 = sigf(o01) * tanhf(c01);
            const float h10 = sigf(o10) * tanhf(c10);
            const float h11 = sigf(o11) * tanhf(c11);
            // store new h (bf16 main + f32 tail) and output
            if (j0 < 192) { hA[j0] = f2bf(h00); hA[192 + j0] = f2bf(h01); }
            else          { ht[j0 - 192] = h00; ht[8 + j0 - 192] = h01; }
            if (j1 < 192) { hA[j1] = f2bf(h10); hA[192 + j1] = f2bf(h11); }
            else          { ht[j1 - 192] = h10; ht[8 + j1 - 192] = h11; }
            const size_t ob = (size_t)t * (BSZ * HID);
            out[ob + j0]       = h00;
            out[ob + HID + j0] = h01;
            out[ob + j1]       = h10;
            out[ob + HID + j1] = h11;
        }
        __syncthreads();          // h writes visible before next step
    }
}

// =====================================================================
extern "C" void kernel_launch(void* const* d_in, const int* in_sizes, int n_in,
                              void* d_out, int out_size, void* d_ws, size_t ws_size,
                              hipStream_t stream) {
    const float* video  = (const float*)d_in[0];  // [2,4096,3,20,20]
    const float* conv_w = (const float*)d_in[1];  // [16,3,3,3]
    const float* W_ih   = (const float*)d_in[2];  // [800,6400]
    const float* W_hh   = (const float*)d_in[3];  // [800,200]
    const float* b_ih   = (const float*)d_in[4];  // [800]
    const float* b_hh   = (const float*)d_in[5];  // [800]
    float* out = (float*)d_out;                   // [4096,2,200]

    char* ws = (char*)d_ws;
    unsigned short* featb = (unsigned short*)ws;                    // 104,857,600 B
    unsigned short* wihb  = (unsigned short*)(ws + 104857600ull);   //  10,240,000 B
    float*          xg    = (float*)(ws + 115097600ull);            //  26,214,400 B

    conv_relu_bf16_kernel<<<BSZ * TLEN, 256, 0, stream>>>(video, conv_w, featb);

    const int nW = NDIM * KDIM;
    cvt_permute_bf16_kernel<<<(nW + 255) / 256, 256, 0, stream>>>(W_ih, wihb);

    dim3 gg(TLEN * BSZ / 256, (NDIM + 63) / 64);  // 32 x 13
    gemm_wmma_kernel<<<gg, 256, 0, stream>>>(featb, wihb, b_ih, b_hh, xg);

    lstm_wmma_kernel<<<1, 800, 0, stream>>>(xg, W_hh, out);
}
